// PreManifestResonanceLayer_60979945668673
// MI455X (gfx1250) — compile-verified, matrix operand order checked by reading, not run on hardware
//
#include <hip/hip_runtime.h>
#include <hip/hip_bf16.h>

typedef __bf16 bf16;
typedef __attribute__((ext_vector_type(16))) __bf16 v16bf;
typedef __attribute__((ext_vector_type(8)))  __bf16 v8bf;
typedef __attribute__((ext_vector_type(8)))  float   v8f;

union BF16x16 {
  v16bf v;
  v8bf  h[2];
  bf16  e[16];
  unsigned int u[8];
};

#define BB 2
#define SS 1024
#define HH 1024
#define VV 50257
#define MM (BB * SS)        // 2048 rows
#define VP 50304            // V padded to a multiple of 64 (786*64)
#define N_ITERS 5
#define EPS_HALT 0.05f

// ---------------------------------------------------------------------------
// CDNA5 async copy global->LDS (ASYNCcnt-tracked), inline asm so it works on
// both ROCm 7.2 and amdgpu-toolchain. One b128 per lane.
// ---------------------------------------------------------------------------
__device__ __forceinline__ void async_copy_b128(void* lds_dst, const void* gsrc) {
  unsigned lds = (unsigned)(size_t)lds_dst;
  unsigned long long g = (unsigned long long)(size_t)gsrc;
  asm volatile("global_load_async_to_lds_b128 %0, %1, off"
               :: "v"(lds), "v"(g) : "memory");
}
#define S_WAIT_ASYNC(imm) asm volatile("s_wait_asynccnt " imm ::: "memory")

__device__ __forceinline__ void zero_frag(BF16x16& f) {
#pragma unroll
  for (int u = 0; u < 8; ++u) f.u[u] = 0u;
}

// ---------------------------------------------------------------------------
// fp32 -> bf16 conversion (grid-stride)
// ---------------------------------------------------------------------------
__global__ void f32_to_bf16_kernel(const float* __restrict__ in,
                                   bf16* __restrict__ out, long n) {
  long i = (long)blockIdx.x * blockDim.x + threadIdx.x;
  const long stride = (long)gridDim.x * blockDim.x;
  for (; i < n; i += stride) out[i] = (bf16)in[i];
}

// ---------------------------------------------------------------------------
// LayerNorm over H=1024 per row + additive-mask row bias
// ---------------------------------------------------------------------------
__global__ __launch_bounds__(256) void ln_mask_kernel(
    const float* __restrict__ x, const float* __restrict__ g,
    const float* __restrict__ b, const float* __restrict__ amask,
    bf16* __restrict__ out, float* __restrict__ rowmask) {
  __shared__ float r1[256];
  __shared__ float r2[256];
  const int row = blockIdx.x;
  const int tid = threadIdx.x;
  const float* xr = x + (size_t)row * HH;
  float vals[4];
  float s = 0.f, sq = 0.f;
#pragma unroll
  for (int j = 0; j < 4; ++j) {
    vals[j] = xr[tid + j * 256];
    s += vals[j];
    sq += vals[j] * vals[j];
  }
  r1[tid] = s; r2[tid] = sq;
  __syncthreads();
  for (int st = 128; st > 0; st >>= 1) {
    if (tid < st) { r1[tid] += r1[tid + st]; r2[tid] += r2[tid + st]; }
    __syncthreads();
  }
  const float mean = r1[0] * (1.0f / HH);
  const float var  = r2[0] * (1.0f / HH) - mean * mean;
  const float rstd = rsqrtf(var + 1e-5f);
  bf16* orow = out + (size_t)row * HH;
#pragma unroll
  for (int j = 0; j < 4; ++j) {
    const int i = tid + j * 256;
    orow[i] = (bf16)((vals[j] - mean) * rstd * g[i] + b[i]);
  }
  if (tid == 0) rowmask[row] = (1.0f - amask[row]) * -10000.0f;
}

// ---------------------------------------------------------------------------
// NT GEMM: C[M,N] = A[M,K] * W[N,K]^T (+biasN[col]) (+biasRow[row])
// Block tile 64M x 256N, K-step 64.  A tile (64x64 bf16, 8KB) async-staged
// into double-buffered LDS (2 async instructions per wave per tile).
// Each wave owns a 64M x 32N strip: 2 N-subtiles x 4 M-subtiles = 8
// accumulators, 16 v_wmma_f32_16x16x32_bf16 per K64 step; each A-fragment
// ds_load feeds two WMMAs.  B fragments register-pipelined one tile ahead.
// ---------------------------------------------------------------------------
__global__ __launch_bounds__(256) void gemm_nt_wmma(
    const bf16* __restrict__ A, const bf16* __restrict__ W,
    const float* __restrict__ biasN, const float* __restrict__ biasRow,
    float* __restrict__ outF, bf16* __restrict__ outB,
    int M, int N, int K, long ldC) {
  __shared__ __align__(16) bf16 As[2][64][72];   // row stride 144B (16B-aligned)

  const int wid  = threadIdx.x >> 5;
  const int lane = threadIdx.x & 31;
  const int l15  = lane & 15;
  const int lhi  = lane >> 4;   // 0 or 1
  const int koff = lhi * 8;     // K-stripe base for this half-wave

  const int m0 = blockIdx.x * 64;
  const int n0 = blockIdx.y * 256 + wid * 32;   // wave strip: [n0, n0+32)

  // staging assignment: each thread copies two 16B chunks of the 64x64 A tile
  const int sr = threadIdx.x >> 2;        // 0..63  tile row
  const int sc = (threadIdx.x & 3) * 8;   // 0,8,16,24

  v8f acc[4][2];
#pragma unroll
  for (int i = 0; i < 4; ++i)
#pragma unroll
    for (int n = 0; n < 2; ++n)
#pragma unroll
      for (int r = 0; r < 8; ++r) acc[i][n][r] = 0.0f;

  const int nrow0 = n0 + l15;
  const int nrow1 = n0 + 16 + l15;
  const bool nok[2] = {nrow0 < N, nrow1 < N};
  const bf16* wrow[2] = {W + (size_t)nrow0 * K, W + (size_t)nrow1 * K};

  // prologue: stage A tile 0, load B fragments for k0=0
  async_copy_b128(&As[0][sr][sc],      A + (size_t)(m0 + sr) * K + sc);
  async_copy_b128(&As[0][sr][sc + 32], A + (size_t)(m0 + sr) * K + sc + 32);

  BF16x16 bcur[2][2];   // [n-subtile][k-substep]
#pragma unroll
  for (int n = 0; n < 2; ++n)
#pragma unroll
    for (int s = 0; s < 2; ++s) {
      if (nok[n]) {
        const bf16* bp = wrow[n] + s * 32 + koff;
        bcur[n][s].h[0] = *(const v8bf*)(bp);
        bcur[n][s].h[1] = *(const v8bf*)(bp + 16);
      } else {
        zero_frag(bcur[n][s]);
      }
    }

  for (int k0 = 0; k0 < K; k0 += 64) {
    const int cur = (k0 >> 6) & 1;
    const bool more = (k0 + 64 < K);

    if (more) {  // stage next A tile into the other buffer
      const bf16* ap = A + (size_t)(m0 + sr) * K + (k0 + 64) + sc;
      async_copy_b128(&As[cur ^ 1][sr][sc],      ap);
      async_copy_b128(&As[cur ^ 1][sr][sc + 32], ap + 32);
      S_WAIT_ASYNC("0x2");           // current tile's 2 copies done; next 2 fly
    } else {
      S_WAIT_ASYNC("0x0");
    }
    __syncthreads();

    // register-pipeline: next B fragments while WMMAs run on bcur
    BF16x16 bnext[2][2];
#pragma unroll
    for (int n = 0; n < 2; ++n)
#pragma unroll
      for (int s = 0; s < 2; ++s) {
        if (more && nok[n]) {
          const bf16* bp = wrow[n] + (k0 + 64) + s * 32 + koff;
          bnext[n][s].h[0] = *(const v8bf*)(bp);
          bnext[n][s].h[1] = *(const v8bf*)(bp + 16);
          if (s == 0) __builtin_prefetch(bp + 128, 0, 3);  // global_prefetch_b8
        } else {
          zero_frag(bnext[n][s]);
        }
      }

#pragma unroll
    for (int s = 0; s < 2; ++s) {
      const int kk = s * 32;
#pragma unroll
      for (int i = 0; i < 4; ++i) {
        BF16x16 afrag;                 // ds_load_b128 x2, reused by 2 WMMAs
        afrag.h[0] = *(const v8bf*)(&As[cur][i * 16 + l15][kk + koff]);
        afrag.h[1] = *(const v8bf*)(&As[cur][i * 16 + l15][kk + koff + 16]);
        acc[i][0] = __builtin_amdgcn_wmma_f32_16x16x32_bf16(
            false, afrag.v, false, bcur[0][s].v, (short)0, acc[i][0], false, false);
        acc[i][1] = __builtin_amdgcn_wmma_f32_16x16x32_bf16(
            false, afrag.v, false, bcur[1][s].v, (short)0, acc[i][1], false, false);
      }
    }
    __syncthreads();                 // tile consumed; safe to overwrite
#pragma unroll
    for (int n = 0; n < 2; ++n)
#pragma unroll
      for (int s = 0; s < 2; ++s) bcur[n][s] = bnext[n][s];
  }

#pragma unroll
  for (int n = 0; n < 2; ++n) {
    const int col = n0 + n * 16 + l15;
    if (col < N) {
      const float bn = biasN ? biasN[col] : 0.0f;
#pragma unroll
      for (int i = 0; i < 4; ++i) {
        const int rbase = m0 + i * 16 + lhi * 8;
#pragma unroll
        for (int r = 0; r < 8; ++r) {
          const int row = rbase + r;
          float val = acc[i][n][r] + bn;
          if (biasRow) val += biasRow[row];
          if (outF) outF[(size_t)row * ldC + col] = val;
          if (outB) outB[(size_t)row * ldC + col] = (bf16)val;
        }
      }
    }
  }
}

// ---------------------------------------------------------------------------
// NN GEMM: C[M,N] = A[M,K] * B[K,N]  (B row-major, rows = K)
// weighted = probs @ E : M=2048, K=V(padded to 64), N=H.
// Block tile 64M x 128N, K-step 64.  A (64x64, 8KB) and B (64x128, 16KB)
// tiles async-staged, double-buffered: 6 async instructions per wave per
// tile.  K-tail: A reads the zeroed pad, B row index clamped (garbage rows
// multiply by zero A columns).  8 WMMAs per K64 step.
// ---------------------------------------------------------------------------
__global__ __launch_bounds__(256) void gemm_nn_wmma(
    const bf16* __restrict__ A, const bf16* __restrict__ Bm,
    float* __restrict__ outF, bf16* __restrict__ outB,
    int M, int N, int K, int Kpad, long ldA, long ldB, long ldC) {
  __shared__ __align__(16) bf16 As[2][64][72];    // 18.4 KB
  __shared__ __align__(16) bf16 Bs[2][64][136];   // 34.8 KB

  const int wid  = threadIdx.x >> 5;
  const int lane = threadIdx.x & 31;
  const int l15  = lane & 15;
  const int lhi  = lane >> 4;
  const int koff = lhi * 8;

  const int m0   = blockIdx.x * 64;
  const int nblk = blockIdx.y * 128;
  const int n0   = nblk + wid * 16;

  // A staging: two 16B chunks per thread (64x64 tile)
  const int ar = threadIdx.x >> 2;        // 0..63
  const int ac = (threadIdx.x & 3) * 8;   // 0,8,16,24
  // B staging: four contiguous 16B chunks per thread (64x128 tile)
  const int br = threadIdx.x >> 2;        // 0..63
  const int bc = (threadIdx.x & 3) * 32;  // 0,32,64,96

  v8f acc[4];
#pragma unroll
  for (int i = 0; i < 4; ++i)
#pragma unroll
    for (int r = 0; r < 8; ++r) acc[i][r] = 0.0f;

  auto stage = [&](int buf, int k0) {
    const bf16* ap = A + (size_t)(m0 + ar) * ldA + k0 + ac;
    async_copy_b128(&As[buf][ar][ac],      ap);
    async_copy_b128(&As[buf][ar][ac + 32], ap + 32);
    int krow = k0 + br;
    if (krow >= K) krow = K - 1;          // clamp: A pad column is zero anyway
    const bf16* bp = Bm + (size_t)krow * ldB + (nblk + bc);
    async_copy_b128(&Bs[buf][br][bc],      bp);
    async_copy_b128(&Bs[buf][br][bc + 8],  bp + 8);
    async_copy_b128(&Bs[buf][br][bc + 16], bp + 16);
    async_copy_b128(&Bs[buf][br][bc + 24], bp + 24);
  };

  stage(0, 0);

  for (int k0 = 0; k0 < Kpad; k0 += 64) {
    const int cur = (k0 >> 6) & 1;
    const bool more = (k0 + 64 < Kpad);

    if (more) {
      stage(cur ^ 1, k0 + 64);
      S_WAIT_ASYNC("0x6");     // current tile's 6 copies retired; next 6 fly
    } else {
      S_WAIT_ASYNC("0x0");
    }
    __syncthreads();

    const int nl = wid * 16 + l15;
#pragma unroll
    for (int s = 0; s < 2; ++s) {
      const int kk = s * 32;
      // gather the K-striped B fragment from LDS
      BF16x16 bfrag;
#pragma unroll
      for (int j = 0; j < 8; ++j) bfrag.e[j]     = Bs[cur][kk + koff + j][nl];
#pragma unroll
      for (int j = 0; j < 8; ++j) bfrag.e[8 + j] = Bs[cur][kk + 16 + koff + j][nl];

#pragma unroll
      for (int i = 0; i < 4; ++i) {
        BF16x16 afrag;
        afrag.h[0] = *(const v8bf*)(&As[cur][i * 16 + l15][kk + koff]);
        afrag.h[1] = *(const v8bf*)(&As[cur][i * 16 + l15][kk + koff + 16]);
        acc[i] = __builtin_amdgcn_wmma_f32_16x16x32_bf16(
            false, afrag.v, false, bfrag.v, (short)0, acc[i], false, false);
      }
    }
    __syncthreads();
  }

  const int col = n0 + l15;
  if (col < N) {
#pragma unroll
    for (int i = 0; i < 4; ++i) {
      const int rbase = m0 + i * 16 + lhi * 8;
#pragma unroll
      for (int r = 0; r < 8; ++r) {
        const int row = rbase + r;
        const float val = acc[i][r];
        if (outF) outF[(size_t)row * ldC + col] = val;
        if (outB) outB[(size_t)row * ldC + col] = (bf16)val;
      }
    }
  }
}

// ---------------------------------------------------------------------------
// Row softmax over V (bf16 scores -> bf16 probs in place) + entropy.
// Gated by `done` flag (flags[0]); contributes to all-halt (flags[1]).
// Zeroes the [V, VP) pad so the NN GEMM's K-tail reads zeros.
// ---------------------------------------------------------------------------
__global__ __launch_bounds__(256) void softmax_entropy_kernel(
    bf16* __restrict__ scores, float* __restrict__ ent_out,
    int* __restrict__ iters_out, int* __restrict__ flags, int t) {
  __shared__ float red[256];
  const int row = blockIdx.x;
  const int tid = threadIdx.x;
  bf16* srow = scores + (size_t)row * VP;

  // pass 1: max
  float mx = -3.0e38f;
  for (int v = tid; v < VV; v += 256) mx = fmaxf(mx, (float)srow[v]);
  red[tid] = mx;
  __syncthreads();
  for (int st = 128; st > 0; st >>= 1) {
    if (tid < st) red[tid] = fmaxf(red[tid], red[tid + st]);
    __syncthreads();
  }
  mx = red[0];
  __syncthreads();

  // pass 2: sum of exp
  float sm = 0.f;
  for (int v = tid; v < VV; v += 256) sm += __expf((float)srow[v] - mx);
  red[tid] = sm;
  __syncthreads();
  for (int st = 128; st > 0; st >>= 1) {
    if (tid < st) red[tid] += red[tid + st];
    __syncthreads();
  }
  const float rinv = 1.0f / red[0];
  __syncthreads();

  // pass 3: write probs, accumulate entropy
  float ent = 0.f;
  for (int v = tid; v < VV; v += 256) {
    const float p = __expf((float)srow[v] - mx) * rinv;
    srow[v] = (bf16)p;
    const float q = p + 1e-10f;
    ent -= q * __logf(q);
  }
  red[tid] = ent;
  __syncthreads();
  for (int st = 128; st > 0; st >>= 1) {
    if (tid < st) red[tid] += red[tid + st];
    __syncthreads();
  }

  // zero the padded tail columns
  for (int v = VV + tid; v < VP; v += 256) srow[v] = (bf16)0.0f;

  if (tid == 0) {
    const float e = red[0];
    if (!flags[0]) {            // not done yet -> update entropy/iters
      ent_out[row] = e;
      iters_out[row] = t + 1;
    }
    atomicAnd(&flags[1], (e < EPS_HALT) ? 1 : 0);
  }
}

// ---------------------------------------------------------------------------
// flag management + query update
// ---------------------------------------------------------------------------
__global__ void init_flags_kernel(int* flags) {
  if (threadIdx.x == 0) flags[0] = 0;
}
__global__ void reset_halt_kernel(int* flags) {
  if (threadIdx.x == 0) flags[1] = 1;
}
__global__ void commit_done_kernel(int* flags) {
  if (threadIdx.x == 0) {
    if (!flags[0] && flags[1]) flags[0] = 1;
  }
}

__global__ void update_query_kernel(float* __restrict__ qf,
                                    bf16* __restrict__ qb,
                                    const float* __restrict__ upd,
                                    const int* __restrict__ flags, long n) {
  const int done = flags[0];   // new_done (commit ran before this kernel)
  long i = (long)blockIdx.x * blockDim.x + threadIdx.x;
  const long stride = (long)gridDim.x * blockDim.x;
  for (; i < n; i += stride) {
    float q = qf[i];
    if (!done) { q += upd[i]; qf[i] = q; }
    qb[i] = (bf16)q;
  }
}

// ---------------------------------------------------------------------------
// launch
// ---------------------------------------------------------------------------
extern "C" void kernel_launch(void* const* d_in, const int* in_sizes, int n_in,
                              void* d_out, int out_size, void* d_ws,
                              size_t ws_size, hipStream_t stream) {
  (void)in_sizes; (void)n_in; (void)out_size; (void)ws_size;

  const float* hidden = (const float*)d_in[0];
  const float* amask  = (const float*)d_in[1];
  const float* Ew     = (const float*)d_in[2];
  const float* Wq     = (const float*)d_in[3];
  const float* bq     = (const float*)d_in[4];
  const float* Wo     = (const float*)d_in[5];
  const float* bo     = (const float*)d_in[6];
  const float* vbias  = (const float*)d_in[7];
  const float* lng    = (const float*)d_in[8];
  const float* lnb    = (const float*)d_in[9];

  float* out    = (float*)d_out;
  float* logits = out;                                   // [M*V]
  float* entO   = out + (size_t)MM * VV;                 // [M]
  int*   itersO = (int*)(out + (size_t)MM * VV + MM);    // [M] int32

  char* ws = (char*)d_ws;
  size_t off = 0;
  auto carve = [&](size_t bytes) -> char* {
    char* p = ws + off;
    off += (bytes + 255) & ~(size_t)255;
    return p;
  };
  bf16*  Ebf   = (bf16*) carve((size_t)VV * HH * 2);   // bf16 embedding (L2-resident)
  bf16*  Wqbf  = (bf16*) carve((size_t)HH * HH * 2);
  bf16*  Wobf  = (bf16*) carve((size_t)HH * HH * 2);
  bf16*  hbf   = (bf16*) carve((size_t)MM * HH * 2);
  float* qf    = (float*)carve((size_t)MM * HH * 4);
  bf16*  qbf   = (bf16*) carve((size_t)MM * HH * 2);
  bf16*  sc    = (bf16*) carve((size_t)MM * VP * 2);   // scores/probs in place
  bf16*  wbf   = (bf16*) carve((size_t)MM * HH * 2);
  float* updf  = (float*)carve((size_t)MM * HH * 4);
  float* rmask = (float*)carve((size_t)MM * 4);
  int*   flags = (int*)  carve(256);

  // weight precision conversion
  f32_to_bf16_kernel<<<4096, 256, 0, stream>>>(Ew, Ebf, (long)VV * HH);
  f32_to_bf16_kernel<<<1024, 256, 0, stream>>>(Wq, Wqbf, (long)HH * HH);
  f32_to_bf16_kernel<<<1024, 256, 0, stream>>>(Wo, Wobf, (long)HH * HH);
  init_flags_kernel<<<1, 32, 0, stream>>>(flags);

  // h = LayerNorm(hidden); rowmask = (1-mask)*-1e4
  ln_mask_kernel<<<MM, 256, 0, stream>>>(hidden, lng, lnb, amask, hbf, rmask);

  // query = h @ Wq^T + bq
  {
    dim3 g(MM / 64, HH / 256);
    gemm_nt_wmma<<<g, 256, 0, stream>>>(hbf, Wqbf, bq, nullptr, qf, qbf,
                                        MM, HH, HH, (long)HH);
  }

  for (int t = 0; t < N_ITERS; ++t) {
    // scores = query @ E^T + vocab_bias + add_mask
    {
      dim3 g(MM / 64, (VV + 255) / 256);
      gemm_nt_wmma<<<g, 256, 0, stream>>>(qbf, Ebf, vbias, rmask, nullptr, sc,
                                          MM, VV, HH, (long)VP);
    }
    reset_halt_kernel<<<1, 32, 0, stream>>>(flags);
    softmax_entropy_kernel<<<MM, 256, 0, stream>>>(sc, entO, itersO, flags, t);
    commit_done_kernel<<<1, 32, 0, stream>>>(flags);
    // weighted = probs @ E
    {
      dim3 g(MM / 64, HH / 128);
      gemm_nn_wmma<<<g, 256, 0, stream>>>(sc, Ebf, nullptr, wbf,
                                          MM, HH, VV, VP,
                                          (long)VP, (long)HH, (long)HH);
    }
    // upd = weighted @ Wo^T + bo
    {
      dim3 g(MM / 64, HH / 256);
      gemm_nt_wmma<<<g, 256, 0, stream>>>(wbf, Wobf, bo, nullptr, updf, nullptr,
                                          MM, HH, HH, (long)HH);
    }
    // query = done ? query : query + upd ; refresh bf16 copy
    update_query_kernel<<<2048, 256, 0, stream>>>(qf, qbf, updf, flags,
                                                  (long)MM * HH);
  }

  // logits = query @ E^T + vocab_bias
  {
    dim3 g(MM / 64, (VV + 255) / 256);
    gemm_nt_wmma<<<g, 256, 0, stream>>>(qbf, Ebf, vbias, nullptr, logits,
                                        nullptr, MM, VV, HH, (long)VV);
  }
}
